// SpatialCrossAttention_15006615734128
// MI455X (gfx1250) — matrix operand order repaired and checked by reading, not compiled, not run on hardware
//
#include <hip/hip_runtime.h>

#define EMBED  256
#define HEADS  8
#define POINTS 8
#define CAMS   6
#define NQ     10000
#define H0     116
#define W0     200
#define NV     (H0 * W0)        // 23200
#define DH     32
#define MROWS  (CAMS * NV)      // 139200
#define NOFF   128              // HEADS*POINTS*2
#define NATT   64               // HEADS*POINTS
#define NCAT   (NOFF + NATT)    // 192

typedef __attribute__((ext_vector_type(16))) __bf16 v16bf;
typedef __attribute__((ext_vector_type(8)))  float  v8f;
typedef __attribute__((ext_vector_type(2)))  float  v2f;

static __device__ __forceinline__ unsigned short f2bf(float x) {
  __bf16 h = (__bf16)x;                       // RNE convert
  return __builtin_bit_cast(unsigned short, h);
}
static __device__ __forceinline__ float bf2f(unsigned short u) {
  unsigned int v = ((unsigned int)u) << 16;   // bf16 -> f32 is exact
  return __builtin_bit_cast(float, v);
}

// ---------------------------------------------------------------------------
// Kernel 0: pre-swizzle w_value (256x256 f32) into per-lane WMMA B-fragment
// layout, bf16:  Wf[((nt*8 + kc)*32 + lane)*16 + i]  where
//   n = nt*16 + (lane&15),  k = kc*32 + (lane>>4)*16 + i
// One thread fills one 16-element fragment (4096 fragments total).
// ---------------------------------------------------------------------------
__global__ __launch_bounds__(256)
void wfrag_kernel(const float* __restrict__ W, __bf16* __restrict__ Wf) {
  const int t    = blockIdx.x * 256 + threadIdx.x;  // 0..4095
  const int nt   = t >> 8;
  const int kc   = (t >> 5) & 7;
  const int lane = t & 31;
  const int n    = nt * 16 + (lane & 15);
  const int kbase = kc * 32 + (lane >> 4) * 16;
  __bf16* dst = Wf + (size_t)t * 16;
#pragma unroll
  for (int i = 0; i < 16; ++i)
    dst[i] = (__bf16)W[(size_t)(kbase + i) * EMBED + n];
}

// ---------------------------------------------------------------------------
// Kernel 1: v_proj = value(139200x256) @ w_value + b_value -> bf16.
// One wave computes a 16x64 strip (4 N-tiles).  Per k-chunk: load A fragment
// + all 4 B fragments first (one load clause), then issue 4 back-to-back
// v_wmma_f32_16x16x32_bf16.  k-loop unrolled x2 for load/WMMA overlap.
// ---------------------------------------------------------------------------
__global__ __launch_bounds__(256)
void value_proj_kernel(const float* __restrict__ V, const v16bf* __restrict__ Wf,
                       const float* __restrict__ bias,
                       unsigned short* __restrict__ out) {
  const int wave  = (blockIdx.x * blockDim.x + threadIdx.x) >> 5;
  const int lane  = threadIdx.x & 31;
  const int mt    = wave >> 2;          // 8700 M tiles
  const int ng    = wave & 3;           // 4 N-groups of 4 tiles
  if (mt * 16 >= MROWS) return;
  const int lhalf = lane >> 4;
  const int l15   = lane & 15;

  const float* arow  = V + (size_t)(mt * 16 + l15) * EMBED;
  const v16bf* wbase = Wf + (size_t)(ng * 4) * 8 * 32 + lane;   // + g*8*32 + kc*32

  v8f acc[4] = {};
#pragma unroll 2
  for (int kc = 0; kc < 8; ++kc) {
    const int k0 = kc * 32;
    __builtin_prefetch(arow + k0 + 64, 0, 0);   // speculative A-stream prefetch
    // ---- load phase: B fragments (4 x 32B) + A (2 x 32B) ----
    v16bf b0 = wbase[(0 * 8 + kc) * 32];
    v16bf b1 = wbase[(1 * 8 + kc) * 32];
    v16bf b2 = wbase[(2 * 8 + kc) * 32];
    v16bf b3 = wbase[(3 * 8 + kc) * 32];
    v16bf a;
#pragma unroll
    for (int i = 0; i < 8; ++i) {       // A 16x32: per-lane K = 16*lhalf + {0..15}
      a[i]     = (__bf16)arow[k0 + lhalf * 16 + i];
      a[i + 8] = (__bf16)arow[k0 + lhalf * 16 + 8 + i];
    }
    // ---- compute phase: 4 back-to-back WMMAs (independent accumulators) ----
    acc[0] = __builtin_amdgcn_wmma_f32_16x16x32_bf16(false, a, false, b0,
                                                     (short)0, acc[0], false, false);
    acc[1] = __builtin_amdgcn_wmma_f32_16x16x32_bf16(false, a, false, b1,
                                                     (short)0, acc[1], false, false);
    acc[2] = __builtin_amdgcn_wmma_f32_16x16x32_bf16(false, a, false, b2,
                                                     (short)0, acc[2], false, false);
    acc[3] = __builtin_amdgcn_wmma_f32_16x16x32_bf16(false, a, false, b3,
                                                     (short)0, acc[3], false, false);
  }
#pragma unroll
  for (int g = 0; g < 4; ++g) {
    const int ncol = (ng * 4 + g) * 16 + l15;
    const float bv = bias[ncol];
#pragma unroll
    for (int r = 0; r < 8; ++r) {       // D: VGPR r -> M = r + 8*lhalf, N = l15
      const int m = mt * 16 + r + lhalf * 8;
      out[(size_t)m * EMBED + ncol] = f2bf(acc[g][r] + bv);
    }
  }
}

// ---------------------------------------------------------------------------
// Kernel 2: offattn = query(10000x256) @ [w_off | w_attn] + bias, f32 out.
// N-tiles 0..7 -> w_off (128 cols), 8..11 -> w_attn (64 cols).
// ---------------------------------------------------------------------------
__global__ __launch_bounds__(256)
void qproj_kernel(const float* __restrict__ Qm,
                  const float* __restrict__ Woff, const float* __restrict__ Boff,
                  const float* __restrict__ Watt, const float* __restrict__ Batt,
                  float* __restrict__ out) {
  const int wave  = (blockIdx.x * blockDim.x + threadIdx.x) >> 5;
  const int lane  = threadIdx.x & 31;
  const int mt    = wave / 12;          // 625 M tiles
  const int nt    = wave % 12;          // 12 N tiles
  if (mt * 16 >= NQ) return;
  const int lhalf = lane >> 4;
  const int l15   = lane & 15;

  const float* B;  const float* bias;  int ldb, nloc;
  if (nt < 8) { B = Woff; bias = Boff; ldb = NOFF; nloc = nt * 16 + l15; }
  else        { B = Watt; bias = Batt; ldb = NATT; nloc = (nt - 8) * 16 + l15; }

  const float* arow = Qm + (size_t)(mt * 16 + l15) * EMBED;

  v8f acc = {};
  for (int k0 = 0; k0 < EMBED; k0 += 32) {
    v16bf a, b;
#pragma unroll
    for (int i = 0; i < 16; ++i) {
      b[i] = (__bf16)B[(size_t)(k0 + lhalf * 16 + i) * ldb + nloc];
    }
#pragma unroll
    for (int i = 0; i < 8; ++i) {
      a[i]     = (__bf16)arow[k0 + lhalf * 16 + i];
      a[i + 8] = (__bf16)arow[k0 + lhalf * 16 + 8 + i];
    }
    acc = __builtin_amdgcn_wmma_f32_16x16x32_bf16(false, a, false, b,
                                                  (short)0, acc, false, false);
  }
  const int ncol = nt * 16 + l15;       // 0..191 in concatenated layout
#pragma unroll
  for (int r = 0; r < 8; ++r) {
    const int m = mt * 16 + r + lhalf * 8;
    out[(size_t)m * NCAT + ncol] = acc[r] + bias[nloc];
  }
}

// ---------------------------------------------------------------------------
// Kernel 3: per-(q,head) softmax over the 8 attn logits (cols 128..191).
// ---------------------------------------------------------------------------
__global__ __launch_bounds__(256)
void softmax_kernel(float* __restrict__ offattn) {
  const int t = blockIdx.x * blockDim.x + threadIdx.x;
  if (t >= NQ * HEADS) return;
  const int q = t >> 3, h = t & 7;
  float* p = offattn + (size_t)q * NCAT + NOFF + h * POINTS;
  float m = p[0];
#pragma unroll
  for (int i = 1; i < POINTS; ++i) m = fmaxf(m, p[i]);
  float s = 0.f, e[POINTS];
#pragma unroll
  for (int i = 0; i < POINTS; ++i) { e[i] = __expf(p[i] - m); s += e[i]; }
  const float inv = 1.f / s;
#pragma unroll
  for (int i = 0; i < POINTS; ++i) p[i] = e[i] * inv;
}

// ---------------------------------------------------------------------------
// Kernel 4: bilinear sampling.  wave = head, lane = dh channel (coalesced 64B
// gathers from bf16 v_proj, L2 resident).  acc[(c,q), h*32+dh].
// ---------------------------------------------------------------------------
#define QPB 40
__global__ __launch_bounds__(256)
void sample_kernel(const unsigned short* __restrict__ vproj,
                   const float* __restrict__ offattn,
                   const float* __restrict__ ref,   // (CAMS,1,NQ,4,2)
                   float* __restrict__ acc) {
  const int c    = blockIdx.x;
  const int h    = threadIdx.x >> 5;
  const int lane = threadIdx.x & 31;
  const unsigned short* vbase = vproj + (size_t)c * NV * EMBED + h * DH + lane;

  for (int qi = 0; qi < QPB; ++qi) {
    const int q = blockIdx.y * QPB + qi;
    const float* refq = ref + ((size_t)c * NQ + q) * 8;            // 4 x (x,y)
    const float* offq = offattn + (size_t)q * NCAT + h * 16;       // 8 x (x,y)
    const float* attq = offattn + (size_t)q * NCAT + NOFF + h * 8; // 8 probs
    float a = 0.f;
#pragma unroll
    for (int p = 0; p < POINTS; ++p) {
      const int d  = p & 3;  // ref point index (POINTS//D x D layout)
      // x = (ref_x + off_x/W0)*W0 - 0.5 = ref_x*W0 + off_x - 0.5 (same for y)
      const float x  = refq[d * 2 + 0] * (float)W0 + offq[p * 2 + 0] - 0.5f;
      const float y  = refq[d * 2 + 1] * (float)H0 + offq[p * 2 + 1] - 0.5f;
      const float aw = attq[p];
      const float x0f = floorf(x), y0f = floorf(y);
      const int   x0  = (int)x0f,  y0  = (int)y0f;
      const float fx  = x - x0f,   fy  = y - y0f;
#pragma unroll
      for (int corner = 0; corner < 4; ++corner) {
        const int dx = corner & 1, dy = corner >> 1;
        const int xc = x0 + dx, yc = y0 + dy;
        const float wc = (dx ? fx : 1.f - fx) * (dy ? fy : 1.f - fy);
        const bool valid = (xc >= 0) && (xc < W0) && (yc >= 0) && (yc < H0);
        const int xcc = min(max(xc, 0), W0 - 1);
        const int ycc = min(max(yc, 0), H0 - 1);
        const float g = bf2f(vbase[(size_t)(ycc * W0 + xcc) * EMBED]);
        a += g * (valid ? wc * aw : 0.f);
      }
    }
    acc[((size_t)c * NQ + q) * EMBED + h * DH + lane] = a;
  }
}

// ---------------------------------------------------------------------------
// Kernel 5: combine cams: slots[q,e] = sum_c hit(c,q)*acc / max(count,1)
// bev_mask is (CAMS,1,NQ,4) bool -> read 4 bytes as one u32 (any != 0).
// ---------------------------------------------------------------------------
__global__ __launch_bounds__(256)
void combine_kernel(const float* __restrict__ acc,
                    const unsigned int* __restrict__ bm,
                    float* __restrict__ slots) {
  const int t = blockIdx.x * blockDim.x + threadIdx.x;
  if (t >= NQ * EMBED) return;
  const int q = t >> 8, e = t & 255;
  float s = 0.f, cnt = 0.f;
#pragma unroll
  for (int c = 0; c < CAMS; ++c) {
    if (bm[c * NQ + q] != 0u) {
      s   += acc[((size_t)c * NQ + q) * EMBED + e];
      cnt += 1.f;
    }
  }
  slots[t] = s / fmaxf(cnt, 1.f);
}

// ---------------------------------------------------------------------------
// Kernel 6: out = slots @ w_out + b_out + query (residual path -> full f32
// precision via V_WMMA_F32_16X16X4_F32).  One wave = 16x64 strip (4 N-tiles);
// per k-step load a + b0..b3 first, then 4 back-to-back WMMAs; unroll x4.
// ---------------------------------------------------------------------------
__global__ __launch_bounds__(256)
void out_proj_kernel(const float* __restrict__ S, const float* __restrict__ W,
                     const float* __restrict__ bias,
                     const float* __restrict__ Qin, float* __restrict__ out) {
  const int wave  = (blockIdx.x * blockDim.x + threadIdx.x) >> 5;
  const int lane  = threadIdx.x & 31;
  const int mt    = wave >> 2;          // 625 M tiles
  const int ng    = wave & 3;           // 4 N-groups of 4 tiles
  if (mt * 16 >= NQ) return;
  const int lhalf = lane >> 4;
  const int l15   = lane & 15;

  const float* arow = S + (size_t)(mt * 16 + l15) * EMBED;
  const int    nc0  = (ng * 4 + 0) * 16 + l15;

  v8f acc[4] = {};
#pragma unroll 4
  for (int k0 = 0; k0 < EMBED; k0 += 4) {
    const float* wrow0 = W + (size_t)(k0 + lhalf * 2 + 0) * EMBED + nc0;
    const float* wrow1 = W + (size_t)(k0 + lhalf * 2 + 1) * EMBED + nc0;
    v2f a, b0, b1, b2, b3;
    a[0]  = arow[k0 + lhalf * 2 + 0];   // A 16x4: lanes<16 K={0,1}, >=16 K={2,3}
    a[1]  = arow[k0 + lhalf * 2 + 1];
    b0[0] = wrow0[0];   b0[1] = wrow1[0];
    b1[0] = wrow0[16];  b1[1] = wrow1[16];
    b2[0] = wrow0[32];  b2[1] = wrow1[32];
    b3[0] = wrow0[48];  b3[1] = wrow1[48];
    acc[0] = __builtin_amdgcn_wmma_f32_16x16x4_f32(false, a, false, b0,
                                                   (short)0, acc[0], false, false);
    acc[1] = __builtin_amdgcn_wmma_f32_16x16x4_f32(false, a, false, b1,
                                                   (short)0, acc[1], false, false);
    acc[2] = __builtin_amdgcn_wmma_f32_16x16x4_f32(false, a, false, b2,
                                                   (short)0, acc[2], false, false);
    acc[3] = __builtin_amdgcn_wmma_f32_16x16x4_f32(false, a, false, b3,
                                                   (short)0, acc[3], false, false);
  }
#pragma unroll
  for (int g = 0; g < 4; ++g) {
    const int ncol = nc0 + g * 16;
    const float bv = bias[ncol];
#pragma unroll
    for (int r = 0; r < 8; ++r) {
      const int m = mt * 16 + r + lhalf * 8;
      out[(size_t)m * EMBED + ncol] = acc[g][r] + bv + Qin[(size_t)m * EMBED + ncol];
    }
  }
}

// ---------------------------------------------------------------------------
extern "C" void kernel_launch(void* const* d_in, const int* in_sizes, int n_in,
                              void* d_out, int out_size, void* d_ws, size_t ws_size,
                              hipStream_t stream) {
  const float* query   = (const float*)d_in[0];
  // d_in[1] = key (unused by reference math)
  const float* value   = (const float*)d_in[2];   // (CAMS,NV,1,EMBED) == flat (MROWS,EMBED)
  const float* ref     = (const float*)d_in[3];
  const unsigned int* bm = (const unsigned int*)d_in[4]; // 4 bool bytes per (c,q)
  const float* w_value = (const float*)d_in[7];
  const float* b_value = (const float*)d_in[8];
  const float* w_off   = (const float*)d_in[9];
  const float* b_off   = (const float*)d_in[10];
  const float* w_attn  = (const float*)d_in[11];
  const float* b_attn  = (const float*)d_in[12];
  const float* w_out   = (const float*)d_in[13];
  const float* b_out   = (const float*)d_in[14];
  float* out = (float*)d_out;

  // workspace layout (all chunks 256B-multiple sized; base from hipMalloc)
  char* p = (char*)d_ws;
  __bf16* wfrag          = (__bf16*)p;          p += (size_t)EMBED * EMBED * 2;     // 128 KB
  unsigned short* vproj  = (unsigned short*)p;  p += (size_t)MROWS * EMBED * 2;     // 71.3 MB
  float* offattn         = (float*)p;           p += (size_t)NQ * NCAT * 4;         //  7.7 MB
  float* accbuf          = (float*)p;           p += (size_t)CAMS * NQ * EMBED * 4; // 61.4 MB
  float* slots           = (float*)p;                                               // 10.2 MB

  // 0) swizzle w_value into bf16 WMMA B-fragments
  wfrag_kernel<<<16, 256, 0, stream>>>(w_value, wfrag);
  // 1) value projection: 8700 M-tiles x 4 N-groups, 8 waves/block
  value_proj_kernel<<<(MROWS / 16) * 4 / 8, 256, 0, stream>>>(value, (const v16bf*)wfrag,
                                                              b_value, vproj);
  // 2) offset+attn projection: 625*12 = 7500 waves
  qproj_kernel<<<(7500 + 7) / 8, 256, 0, stream>>>(query, w_off, b_off, w_attn, b_attn, offattn);
  // 3) softmax over 8 points per (q, head)
  softmax_kernel<<<(NQ * HEADS + 255) / 256, 256, 0, stream>>>(offattn);
  // 4) bilinear sampling: (cam, 250 query blocks) x 256 threads
  sample_kernel<<<dim3(CAMS, NQ / QPB), 256, 0, stream>>>(vproj, offattn, ref, accbuf);
  // 5) cam combine
  combine_kernel<<<(NQ * EMBED + 255) / 256, 256, 0, stream>>>(accbuf, bm, slots);
  // 6) output projection + residual (f32 WMMA), 625 M-tiles x 4 N-groups
  out_proj_kernel<<<(625 * 4 + 7) / 8, 256, 0, stream>>>(slots, w_out, b_out, query, out);
}
